// HybridAIDetector_11613591568830
// MI455X (gfx1250) — compile-verified, matrix operand order checked by reading, not run on hardware
//
#include <hip/hip_runtime.h>
#include <hip/hip_bf16.h>

// HybridAIDetector for MI455X (gfx1250, wave32).
// - All GEMM-shaped work (conv2/conv3 implicit-im2col, every MLP layer) runs on
//   v_wmma_f32_16x16x32_bf16 with f32 accumulation.
// - Register tiling: conv waves reuse one gathered im2col B-fragment across
//   4-8 output-channel WMMAs; GEMM waves reuse one A-fragment across 2-4 column
//   tiles (amortizes the expensive operand over many matrix ops).
// - FFT (512-pt radix-2 in LDS, two passes with transposed writes), edge/noise
//   statistics and conv1 are fused f32 VALU kernels (bandwidth bound anyway).
// - Deterministic: integer atomics for the histogram, fixed-order tree
//   reductions for all float sums.

typedef __attribute__((ext_vector_type(16))) __bf16 v16bf;
typedef __attribute__((ext_vector_type(8)))  float  v8f;

#define BATCH 32
#define HW    512
#define NPIX  (HW*HW)

__device__ __forceinline__ __bf16 f2bf(float f){ return (__bf16)f; }
__device__ __forceinline__ int refl(int i){ i = i < 0 ? -i : i; return i > (HW-1) ? (2*HW-2) - i : i; }

// ---------------------------------------------------------------------------
// WMMA fragment loader.  Layout per CDNA5 ISA 7.12.2 (16-bit A 16x32):
//   lane L holds row r0+(L&15); lanes 0-15 carry K = {k0+0..7, k0+16..23},
//   lanes 16-31 carry K = {k0+8..15, k0+24..31}.  Two contiguous 8xbf16 runs
//   -> compiler emits two global_load_b128 per fragment.
// Same pattern serves the B fragment when weights are stored transposed [N][K].
// ---------------------------------------------------------------------------
__device__ __forceinline__ v16bf load_frag(const __bf16* P, int ld, int r0, int k0, int lane){
  int r  = r0 + (lane & 15);
  int hl = lane >> 4;
  const __bf16* p = P + (size_t)r * ld + k0 + hl * 8;
  v16bf v;
#pragma unroll
  for (int e = 0; e < 8; ++e) v[e]     = p[e];
#pragma unroll
  for (int e = 0; e < 8; ++e) v[8 + e] = p[16 + e];
  return v;
}

// ---------------------------------------------------------------------------
// Generic WMMA GEMM:  D[M,N] = act(A[M,K] * W[K,N] + bias),  A bf16 [M,lda],
// W stored transposed bf16 [N,ldw] (ldw = padded K).  Each wave computes a
// 16 x (16*NT) strip: one A fragment is reused across NT column tiles.
// Writes bf16 into D at column offset coloff with row stride ldd (lets branch
// outputs land directly inside the 512-wide concat buffer).
// ---------------------------------------------------------------------------
template<int NT>
__global__ void wmma_gemm(const __bf16* __restrict__ A, int lda,
                          const __bf16* __restrict__ WT, int ldw,
                          const float* __restrict__ bias,
                          __bf16* __restrict__ D, int ldd, int coloff,
                          int K, int relu)
{
  int lane = threadIdx.x;
  int col0 = blockIdx.x * 16 * NT;
  int row0 = blockIdx.y * 16;
  v8f acc[NT];
#pragma unroll
  for (int t = 0; t < NT; ++t) acc[t] = (v8f){};
  for (int k0 = 0; k0 < K; k0 += 32){
    __builtin_prefetch(WT + (size_t)col0 * ldw + k0 + 64, 0, 0);
    v16bf a = load_frag(A, lda, row0, k0, lane);
#pragma unroll
    for (int t = 0; t < NT; ++t){
      v16bf b = load_frag(WT, ldw, col0 + 16 * t, k0, lane);
      acc[t] = __builtin_amdgcn_wmma_f32_16x16x32_bf16(false, a, false, b, (short)0, acc[t], false, false);
    }
  }
  int mo = (lane >> 4) * 8;          // C/D layout: lanes>=16 hold M = r+8
#pragma unroll
  for (int t = 0; t < NT; ++t){
    int n  = col0 + 16 * t + (lane & 15);
    float bz = bias ? bias[n] : 0.f;
#pragma unroll
    for (int r = 0; r < 8; ++r){
      float v = acc[t][r] + bz;
      if (relu) v = fmaxf(v, 0.f);
      D[(size_t)(row0 + mo + r) * ldd + coloff + n] = f2bf(v);
    }
  }
}

// ---------------------------------------------------------------------------
// Implicit-im2col conv via WMMA: 3x3, stride 2, pad 1.
// A = weights bf16 [OUTC][KTOT] (natural OIHW flatten).  The gathered B
// fragment (16 bounds-checked loads/lane) is built ONCE per K-step and reused
// across OCT output-channel WMMAs (OCT=4 -> 64ch, OCT=8 -> 128ch per wave).
// grid = (pixTiles, OUTC/(16*OCT), B), block = 32 (one wave).
// ---------------------------------------------------------------------------
template<int INC, int INH, int OUTC, int OUTH, int KTOT, int OCT>
__global__ void conv_wmma(const float* __restrict__ in, const __bf16* __restrict__ Wb,
                          const float* __restrict__ bias, float* __restrict__ out)
{
  int lane = threadIdx.x;
  int b    = blockIdx.z;
  int oc0  = blockIdx.y * 16 * OCT;
  int pix0 = blockIdx.x * 16;
  int myPix = pix0 + (lane & 15);
  int py = myPix / OUTH, px = myPix % OUTH;
  int hl = lane >> 4;
  const float* inB = in + (size_t)b * INC * INH * INH;
  v8f acc[OCT];
#pragma unroll
  for (int t = 0; t < OCT; ++t) acc[t] = (v8f){};
  for (int k0 = 0; k0 < KTOT; k0 += 32){
    v16bf bf;
#pragma unroll
    for (int e = 0; e < 16; ++e){
      int k  = k0 + (e >> 3) * 16 + hl * 8 + (e & 7);
      int c  = k / 9, t = k % 9;
      int ky = t / 3, kx = t % 3;
      int iy = py * 2 - 1 + ky, ix = px * 2 - 1 + kx;
      bool ok = (iy >= 0) & (iy < INH) & (ix >= 0) & (ix < INH);
      float v = ok ? inB[((size_t)c * INH + iy) * INH + ix] : 0.f;
      bf[e] = f2bf(v);
    }
#pragma unroll
    for (int t = 0; t < OCT; ++t){
      v16bf a = load_frag(Wb, KTOT, oc0 + 16 * t, k0, lane);
      acc[t] = __builtin_amdgcn_wmma_f32_16x16x32_bf16(false, a, false, bf, (short)0, acc[t], false, false);
    }
  }
  int mo = hl * 8;
  float* outB = out + (size_t)b * OUTC * OUTH * OUTH;
#pragma unroll
  for (int t = 0; t < OCT; ++t){
#pragma unroll
    for (int r = 0; r < 8; ++r){
      int oc = oc0 + 16 * t + mo + r;
      outB[(size_t)oc * OUTH * OUTH + myPix] = acc[t][r] + bias[oc];
    }
  }
}

// ---------------------------------------------------------------------------
// Weight packing
// ---------------------------------------------------------------------------
__global__ void pack_wT(const float* __restrict__ src, __bf16* __restrict__ dst,
                        int K, int N, int Kp)
{
  int i = blockIdx.x * blockDim.x + threadIdx.x;
  if (i >= N * Kp) return;
  int n = i / Kp, k = i % Kp;
  dst[i] = f2bf(k < K ? src[(size_t)k * N + n] : 0.f);
}

__global__ void cvt_f32_bf16(const float* __restrict__ s, __bf16* __restrict__ d, int n){
  int i = blockIdx.x * blockDim.x + threadIdx.x;
  if (i < n) d[i] = f2bf(s[i]);
}

__global__ void zero_u32(unsigned int* p, int n){
  int i = blockIdx.x * blockDim.x + threadIdx.x;
  if (i < n) p[i] = 0u;
}

// ---------------------------------------------------------------------------
// Grayscale + per-channel partial sums (spatial branch means)
// grid (1024, B), block 256
// ---------------------------------------------------------------------------
__global__ void gray_kernel(const float* __restrict__ x, float* __restrict__ gray,
                            float* __restrict__ grayP)
{
  __shared__ float red[256];
  int b = blockIdx.y;
  int p = blockIdx.x * 256 + threadIdx.x;
  const float* xb = x + (size_t)b * 3 * NPIX;
  float r = xb[p], g = xb[NPIX + p], bl = xb[2 * NPIX + p];
  gray[(size_t)b * NPIX + p] = 0.299f * r + 0.587f * g + 0.114f * bl;
  float vals[3] = { r, g, bl };
  for (int c = 0; c < 3; ++c){
    red[threadIdx.x] = vals[c]; __syncthreads();
    for (int s = 128; s > 0; s >>= 1){
      if (threadIdx.x < s) red[threadIdx.x] += red[threadIdx.x + s];
      __syncthreads();
    }
    if (threadIdx.x == 0) grayP[((size_t)b * 1024 + blockIdx.x) * 3 + c] = red[0];
    __syncthreads();
  }
}

__global__ void finalize_chan(const float* __restrict__ grayP, __bf16* __restrict__ A0)
{
  __shared__ float red[256];
  __shared__ float tot[3];
  int b = blockIdx.x;
  float s[3] = {0.f, 0.f, 0.f};
  for (int i = threadIdx.x; i < 1024; i += 256)
    for (int c = 0; c < 3; ++c) s[c] += grayP[((size_t)b * 1024 + i) * 3 + c];
  for (int c = 0; c < 3; ++c){
    red[threadIdx.x] = s[c]; __syncthreads();
    for (int st = 128; st > 0; st >>= 1){
      if (threadIdx.x < st) red[threadIdx.x] += red[threadIdx.x + st];
      __syncthreads();
    }
    if (threadIdx.x == 0) tot[c] = red[0];
    __syncthreads();
  }
  if (threadIdx.x == 0)
    for (int i = 0; i < 32; ++i)
      A0[b * 32 + i] = f2bf(i < 3 ? tot[i] / (float)NPIX : 0.f);
}

// ---------------------------------------------------------------------------
// 512-point radix-2 DIT FFT in LDS; writes transposed (two passes = full 2D).
// grid (512, B), block 256
// ---------------------------------------------------------------------------
__device__ __forceinline__ int bitrev9(int i){
  int r = 0;
#pragma unroll
  for (int k = 0; k < 9; ++k) r = (r << 1) | ((i >> k) & 1);
  return r;
}

__global__ void fft512(const float* __restrict__ rin, const float2* __restrict__ cin,
                       float2* __restrict__ cout, int isReal)
{
  __shared__ float sre[512], sim[512];
  int t = threadIdx.x;
  int row = blockIdx.x;
  int b = blockIdx.y;
  size_t base = ((size_t)b * 512 + row) * 512;
#pragma unroll
  for (int j = 0; j < 2; ++j){
    int i = t + j * 256;
    int s = bitrev9(i);
    if (isReal){ sre[s] = rin[base + i]; sim[s] = 0.f; }
    else { float2 v = cin[base + i]; sre[s] = v.x; sim[s] = v.y; }
  }
  __syncthreads();
  for (int st = 0; st < 9; ++st){
    int hlf = 1 << st;
    int j   = t & (hlf - 1);
    int i0  = ((t >> st) << (st + 1)) + j;
    int i1  = i0 + hlf;
    float ang = -3.14159265358979323846f * (float)j / (float)hlf;
    float sw, cw; __sincosf(ang, &sw, &cw);
    float ar = sre[i0], ai = sim[i0], br = sre[i1], bi = sim[i1];
    float tr = br * cw - bi * sw;
    float ti = br * sw + bi * cw;
    sre[i0] = ar + tr; sim[i0] = ai + ti;
    sre[i1] = ar - tr; sim[i1] = ai - ti;
    __syncthreads();
  }
  size_t ob = (size_t)b * NPIX;
#pragma unroll
  for (int j = 0; j < 2; ++j){
    int i = t + j * 256;
    cout[ob + (size_t)i * 512 + row] = make_float2(sre[i], sim[i]);  // transposed
  }
}

// log1p(|fftshift(F)|) + partial mean/var sums.  grid (1024,B), block 256
__global__ void lm_kernel(const float2* __restrict__ F, float* __restrict__ lm,
                          float* __restrict__ lmP)
{
  __shared__ float red[256];
  int b = blockIdx.y;
  int p = blockIdx.x * 256 + threadIdx.x;
  int y = p >> 9, x = p & 511;
  int ys = (y + 256) & 511, xs = (x + 256) & 511;
  float2 v = F[(size_t)b * NPIX + (size_t)ys * 512 + xs];
  float m = log1pf(sqrtf(v.x * v.x + v.y * v.y));
  lm[(size_t)b * NPIX + p] = m;
  float vals[2] = { m, m * m };
  for (int c = 0; c < 2; ++c){
    red[threadIdx.x] = vals[c]; __syncthreads();
    for (int s = 128; s > 0; s >>= 1){
      if (threadIdx.x < s) red[threadIdx.x] += red[threadIdx.x + s];
      __syncthreads();
    }
    if (threadIdx.x == 0) lmP[((size_t)b * 1024 + blockIdx.x) * 2 + c] = red[0];
    __syncthreads();
  }
}

__global__ void finalize_lm(const float* __restrict__ lmP, float* __restrict__ stats)
{
  __shared__ float red[256];
  __shared__ float tot[2];
  int b = blockIdx.x;
  float s0 = 0.f, s1 = 0.f;
  for (int i = threadIdx.x; i < 1024; i += 256){
    s0 += lmP[((size_t)b * 1024 + i) * 2];
    s1 += lmP[((size_t)b * 1024 + i) * 2 + 1];
  }
  float s[2] = { s0, s1 };
  for (int c = 0; c < 2; ++c){
    red[threadIdx.x] = s[c]; __syncthreads();
    for (int st = 128; st > 0; st >>= 1){
      if (threadIdx.x < st) red[threadIdx.x] += red[threadIdx.x + st];
      __syncthreads();
    }
    if (threadIdx.x == 0) tot[c] = red[0];
    __syncthreads();
  }
  if (threadIdx.x == 0){
    float mu  = tot[0] / (float)NPIX;
    float var = tot[1] / (float)NPIX - mu * mu;
    stats[b * 2]     = mu;
    stats[b * 2 + 1] = sqrtf(fmaxf(var, 0.f));
  }
}

__global__ void norm_lm(float* __restrict__ lm, const float* __restrict__ stats)
{
  int b = blockIdx.y;
  int p = blockIdx.x * 256 + threadIdx.x;
  float mu = stats[b * 2], sd = stats[b * 2 + 1];
  size_t i = (size_t)b * NPIX + p;
  lm[i] = (lm[i] - mu) / (sd + 1e-8f);
}

// ---------------------------------------------------------------------------
// conv1 (1->32, 5x5, s2, zero-pad) + BN + ReLU + 2x2 maxpool fused.
// grid (64 tiles, 32 oc, B), block (16,16); one pooled pixel per thread.
// ---------------------------------------------------------------------------
__global__ void conv1_fused(const float* __restrict__ f, const float* __restrict__ w,
                            const float* __restrict__ bias, const float* __restrict__ g,
                            const float* __restrict__ bb, float* __restrict__ out)
{
  __shared__ float wk[25];
  int oc = blockIdx.y, b = blockIdx.z;
  int tile = blockIdx.x;
  int py = (tile >> 3) * 16 + threadIdx.y;
  int px = (tile & 7)  * 16 + threadIdx.x;
  int tid = threadIdx.y * 16 + threadIdx.x;
  if (tid < 25) wk[tid] = w[oc * 25 + tid];
  __syncthreads();
  float scale = g[oc] * rsqrtf(1.0f + 1e-5f);
  float shift = bb[oc];
  float bs = bias[oc];
  const float* fb = f + (size_t)b * NPIX;
  float best = -1e30f;
#pragma unroll
  for (int sy = 0; sy < 2; ++sy)
#pragma unroll
  for (int sx = 0; sx < 2; ++sx){
    int cy = py * 2 + sy, cx = px * 2 + sx;   // conv output coords (0..255)
    float acc = bs;
#pragma unroll
    for (int dy = 0; dy < 5; ++dy){
      int iy = cy * 2 - 2 + dy;
      if (iy < 0 || iy >= HW) continue;
#pragma unroll
      for (int dx = 0; dx < 5; ++dx){
        int ix = cx * 2 - 2 + dx;
        if (ix < 0 || ix >= HW) continue;
        acc += fb[(size_t)iy * HW + ix] * wk[dy * 5 + dx];
      }
    }
    best = fmaxf(best, fmaxf(acc * scale + shift, 0.f));
  }
  out[(((size_t)b * 32 + oc) * 128 + py) * 128 + px] = best;
}

// BN + ReLU + 2x2 maxpool (conv bias already added).
__global__ void bn_relu_pool(const float* __restrict__ in, float* __restrict__ out,
                             const float* __restrict__ g, const float* __restrict__ bb,
                             int C, int Hin)
{
  int Ho = Hin / 2;
  int idx = blockIdx.x * 256 + threadIdx.x;
  int total = BATCH * C * Ho * Ho;
  if (idx >= total) return;
  int px = idx % Ho, py = (idx / Ho) % Ho, c = (idx / (Ho * Ho)) % C, b = idx / (Ho * Ho * C);
  float scale = g[c] * rsqrtf(1.0f + 1e-5f), shift = bb[c];
  const float* ib = in + ((size_t)b * C + c) * Hin * Hin;
  float m = -1e30f;
  for (int sy = 0; sy < 2; ++sy)
    for (int sx = 0; sx < 2; ++sx)
      m = fmaxf(m, ib[(size_t)(py * 2 + sy) * Hin + px * 2 + sx] * scale + shift);
  out[idx] = fmaxf(m, 0.f);
}

// conv3 BN + ReLU + global mean over 16x16 -> bf16 feature row. grid (B,128)
__global__ void conv3_bn_mean(const float* __restrict__ in, const float* __restrict__ g,
                              const float* __restrict__ bb, __bf16* __restrict__ hfeatB)
{
  __shared__ float red[256];
  int b = blockIdx.x, c = blockIdx.y;
  float scale = g[c] * rsqrtf(1.0f + 1e-5f), shift = bb[c];
  float v = in[((size_t)b * 128 + c) * 256 + threadIdx.x] * scale + shift;
  red[threadIdx.x] = fmaxf(v, 0.f);
  __syncthreads();
  for (int s = 128; s > 0; s >>= 1){
    if (threadIdx.x < s) red[threadIdx.x] += red[threadIdx.x + s];
    __syncthreads();
  }
  if (threadIdx.x == 0) hfeatB[(size_t)b * 128 + c] = f2bf(red[0] / 256.f);
}

// ---------------------------------------------------------------------------
// Edge features: Sobel / threshold / dilation / angle & Laplacian variance.
// grid (32,32,B), block (16,16); 20x20 reflect-101 tile in LDS.
// ---------------------------------------------------------------------------
__global__ void edge_kernel(const float* __restrict__ gray, float* __restrict__ eP)
{
  __shared__ float tile[20][20];
  __shared__ float red[256];
  int b = blockIdx.z;
  int by = blockIdx.y * 16, bx = blockIdx.x * 16;
  int tid = threadIdx.y * 16 + threadIdx.x;
  const float* gb = gray + (size_t)b * NPIX;
  for (int i = tid; i < 400; i += 256){
    int ty = i / 20, tx = i % 20;
    tile[ty][tx] = 255.f * gb[(size_t)refl(by + ty - 2) * HW + refl(bx + tx - 2)];
  }
  __syncthreads();
  int ly = threadIdx.y + 2, lx = threadIdx.x + 2;
  auto sob = [&](int cy, int cx, float& sx, float& sy){
    float a00 = tile[cy-1][cx-1], a01 = tile[cy-1][cx], a02 = tile[cy-1][cx+1];
    float a10 = tile[cy][cx-1],                         a12 = tile[cy][cx+1];
    float a20 = tile[cy+1][cx-1], a21 = tile[cy+1][cx], a22 = tile[cy+1][cx+1];
    sx = -a00 + a02 - 2.f*a10 + 2.f*a12 - a20 + a22;
    sy = -a00 - 2.f*a01 - a02 + a20 + 2.f*a21 + a22;
  };
  float sx, sy; sob(ly, lx, sx, sy);
  float mag  = sqrtf(sx*sx + sy*sy);
  float edge = mag > 150.f ? 1.f : 0.f;
  float ang  = atan2f(sy, sx);
  float lap  = tile[ly-1][lx] + tile[ly+1][lx] + tile[ly][lx-1] + tile[ly][lx+1] - 4.f*tile[ly][lx];
  float dil  = 0.f;
  int gy = by + threadIdx.y, gx = bx + threadIdx.x;
  for (int dy = -1; dy <= 1; ++dy)
    for (int dx = -1; dx <= 1; ++dx){
      int qy = gy + dy, qx = gx + dx;
      if (qy < 0 || qy >= HW || qx < 0 || qx >= HW) continue;
      float qsx, qsy; sob(ly + dy, lx + dx, qsx, qsy);
      if (qsx*qsx + qsy*qsy > 150.f*150.f) dil = 1.f;
    }
  float vals[7] = { edge, mag, dil, ang, ang*ang, lap, lap*lap };
  int blk = blockIdx.y * 32 + blockIdx.x;
  for (int c = 0; c < 7; ++c){
    red[tid] = vals[c]; __syncthreads();
    for (int s = 128; s > 0; s >>= 1){
      if (tid < s) red[tid] += red[tid + s];
      __syncthreads();
    }
    if (tid == 0) eP[((size_t)b * 1024 + blk) * 7 + c] = red[0];
    __syncthreads();
  }
}

__global__ void finalize_edge(const float* __restrict__ eP, __bf16* __restrict__ efeatB)
{
  __shared__ float red[256];
  __shared__ float tot[7];
  int b = blockIdx.x;
  float s[7] = {0,0,0,0,0,0,0};
  for (int i = threadIdx.x; i < 1024; i += 256)
    for (int c = 0; c < 7; ++c) s[c] += eP[((size_t)b * 1024 + i) * 7 + c];
  for (int c = 0; c < 7; ++c){
    red[threadIdx.x] = s[c]; __syncthreads();
    for (int st = 128; st > 0; st >>= 1){
      if (threadIdx.x < st) red[threadIdx.x] += red[threadIdx.x + st];
      __syncthreads();
    }
    if (threadIdx.x == 0) tot[c] = red[0];
    __syncthreads();
  }
  if (threadIdx.x == 0){
    const float N = (float)NPIX;
    float density    = tot[0] / N;
    float strength   = tot[1] / N / 255.f;
    float continuity = tot[2] / (tot[0] + 1e-10f);
    float am = tot[3] / N;
    float ang_var = tot[4] / N - am * am;
    float lmn = tot[5] / N;
    float hf  = (tot[6] / N - lmn * lmn) / 1000.f;
    float f[5] = { density, strength, continuity, ang_var, hf };
    for (int i = 0; i < 32; ++i) efeatB[b * 32 + i] = f2bf(i < 5 ? f[i] : 0.f);
  }
}

// ---------------------------------------------------------------------------
// Noise features: Gaussian(5x5,sigma=1) residual of gray + 3 channels.
// grid (32,32,B), block (16,16). Histogram via integer atomics (deterministic).
// ---------------------------------------------------------------------------
__global__ void noise_kernel(const float* __restrict__ x, const float* __restrict__ gray,
                             float* __restrict__ nP, unsigned int* __restrict__ hist)
{
  __shared__ float tg[400];
  __shared__ float tc[3][400];
  __shared__ float red[256];
  __shared__ unsigned int lh[50];
  int b = blockIdx.z;
  int by = blockIdx.y * 16, bx = blockIdx.x * 16;
  int tid = threadIdx.y * 16 + threadIdx.x;
  if (tid < 50) lh[tid] = 0u;
  const float* gb = gray + (size_t)b * NPIX;
  const float* xb = x + (size_t)b * 3 * NPIX;
  for (int i = tid; i < 400; i += 256){
    int ty = i / 20, tx = i % 20;
    size_t gi = (size_t)refl(by + ty - 2) * HW + refl(bx + tx - 2);
    tg[i]    = gb[gi];
    tc[0][i] = xb[gi];
    tc[1][i] = xb[NPIX + gi];
    tc[2][i] = xb[2 * NPIX + gi];
  }
  __syncthreads();
  const float w5[5] = { 0.054488685f, 0.244201342f, 0.402619946f, 0.244201342f, 0.054488685f };
  int ly = threadIdx.y + 2, lx = threadIdx.x + 2;
  float bg = 0.f, b0 = 0.f, b1 = 0.f, b2 = 0.f;
  for (int dy = 0; dy < 5; ++dy){
    float wy = w5[dy];
    for (int dx = 0; dx < 5; ++dx){
      float w = wy * w5[dx];
      int ii = (ly - 2 + dy) * 20 + (lx - 2 + dx);
      bg += w * tg[ii]; b0 += w * tc[0][ii]; b1 += w * tc[1][ii]; b2 += w * tc[2][ii];
    }
  }
  int ctr = ly * 20 + lx;
  float n  = tg[ctr] - bg;
  float r0 = tc[0][ctr] - b0, r1 = tc[1][ctr] - b1, r2 = tc[2][ctr] - b2;
  if (n >= -0.5f && n <= 0.5f){
    int bin = (int)floorf((n + 0.5f) * 50.f);
    bin = bin < 0 ? 0 : (bin > 49 ? 49 : bin);
    atomicAdd(&lh[bin], 1u);
  }
  float vals[8] = { n, n*n, r0, r0*r0, r1, r1*r1, r2, r2*r2 };
  int blk = blockIdx.y * 32 + blockIdx.x;
  for (int c = 0; c < 8; ++c){
    red[tid] = vals[c]; __syncthreads();
    for (int s = 128; s > 0; s >>= 1){
      if (tid < s) red[tid] += red[tid + s];
      __syncthreads();
    }
    if (tid == 0) nP[((size_t)b * 1024 + blk) * 8 + c] = red[0];
    __syncthreads();
  }
  if (tid < 50) atomicAdd(&hist[b * 50 + tid], lh[tid]);
}

__global__ void finalize_noise(const float* __restrict__ nP, const unsigned int* __restrict__ hist,
                               __bf16* __restrict__ nfeatB)
{
  __shared__ float red[256];
  __shared__ float tot[8];
  int b = blockIdx.x;
  float s[8] = {0,0,0,0,0,0,0,0};
  for (int i = threadIdx.x; i < 1024; i += 256)
    for (int c = 0; c < 8; ++c) s[c] += nP[((size_t)b * 1024 + i) * 8 + c];
  for (int c = 0; c < 8; ++c){
    red[threadIdx.x] = s[c]; __syncthreads();
    for (int st = 128; st > 0; st >>= 1){
      if (threadIdx.x < st) red[threadIdx.x] += red[threadIdx.x + st];
      __syncthreads();
    }
    if (threadIdx.x == 0) tot[c] = red[0];
    __syncthreads();
  }
  if (threadIdx.x == 0){
    const float N = (float)NPIX;
    float mu = tot[0] / N;
    float lv = tot[1] / N - mu * mu;
    float sd = sqrtf(fmaxf(lv, 0.f));
    float tsum = 0.f;
    for (int i = 0; i < 50; ++i) tsum += (float)hist[b * 50 + i];
    float ent = 0.f;
    for (int i = 0; i < 50; ++i){
      float p = (float)hist[b * 50 + i] / (tsum + 1e-10f);
      ent -= p * log2f(p + 1e-10f);
    }
    float cn[3];
    for (int c = 0; c < 3; ++c){
      float m = tot[2 + 2*c] / N;
      cn[c] = sqrtf(fmaxf(tot[3 + 2*c] / N - m * m, 0.f));
    }
    float f[6] = { lv, sd, ent, cn[0], cn[1], cn[2] };
    for (int i = 0; i < 32; ++i) nfeatB[b * 32 + i] = f2bf(i < 6 ? f[i] : 0.f);
  }
}

// Final 128 -> 2 layer (tiny; VALU).  <<<1,64>>>
__global__ void final_logits(const __bf16* __restrict__ act, const float* __restrict__ w3,
                             const float* __restrict__ b3, float* __restrict__ out)
{
  int t = threadIdx.x;
  if (t >= 64) return;
  int b = t >> 1, j = t & 1;
  float s = b3[j];
  for (int k = 0; k < 128; ++k) s += (float)act[b * 128 + k] * w3[k * 2 + j];
  out[b * 2 + j] = s;
}

// ---------------------------------------------------------------------------
// Host launch
// ---------------------------------------------------------------------------
static inline int cdiv(int a, int b){ return (a + b - 1) / b; }

extern "C" void kernel_launch(void* const* d_in, const int* in_sizes, int n_in,
                              void* d_out, int out_size, void* d_ws, size_t ws_size,
                              hipStream_t stream)
{
  (void)in_sizes; (void)n_in; (void)out_size; (void)ws_size;
  const float* x       = (const float*)d_in[0];
  const float* bb_w    = (const float*)d_in[1];
  const float* bb_b    = (const float*)d_in[2];
  const float* proj_w1 = (const float*)d_in[3];
  const float* proj_b1 = (const float*)d_in[4];
  const float* proj_w2 = (const float*)d_in[5];
  const float* proj_b2 = (const float*)d_in[6];
  const float* c1_w    = (const float*)d_in[7];
  const float* c1_b    = (const float*)d_in[8];
  const float* bn1_g   = (const float*)d_in[9];
  const float* bn1_b   = (const float*)d_in[10];
  const float* c2_w    = (const float*)d_in[11];
  const float* c2_b    = (const float*)d_in[12];
  const float* bn2_g   = (const float*)d_in[13];
  const float* bn2_b   = (const float*)d_in[14];
  const float* c3_w    = (const float*)d_in[15];
  const float* c3_b    = (const float*)d_in[16];
  const float* bn3_g   = (const float*)d_in[17];
  const float* bn3_b   = (const float*)d_in[18];
  const float* f_w1    = (const float*)d_in[19];
  const float* f_b1    = (const float*)d_in[20];
  const float* f_w2    = (const float*)d_in[21];
  const float* f_b2    = (const float*)d_in[22];
  const float* n_w1    = (const float*)d_in[23];
  const float* n_b1    = (const float*)d_in[24];
  const float* n_w2    = (const float*)d_in[25];
  const float* n_b2    = (const float*)d_in[26];
  const float* e_w1    = (const float*)d_in[27];
  const float* e_b1    = (const float*)d_in[28];
  const float* e_w2    = (const float*)d_in[29];
  const float* e_b2    = (const float*)d_in[30];
  const float* fu_w1   = (const float*)d_in[31];
  const float* fu_b1   = (const float*)d_in[32];
  const float* fu_w2   = (const float*)d_in[33];
  const float* fu_b2   = (const float*)d_in[34];
  const float* fu_w3   = (const float*)d_in[35];
  const float* fu_b3   = (const float*)d_in[36];
  float* out = (float*)d_out;

  // ---- workspace layout (~220 MB; FFT buffers aliased for lm / h1) ----
  char* W = (char*)d_ws;
  size_t off = 0;
  auto alloc = [&](size_t bytes){ void* p = W + off; off += (bytes + 255) & ~(size_t)255; return p; };
  const size_t SZ_GRAY = (size_t)BATCH * NPIX * sizeof(float);
  const size_t SZ_C    = (size_t)BATCH * NPIX * sizeof(float2);
  float*  gray = (float*) alloc(SZ_GRAY);
  float2* cA   = (float2*)alloc(SZ_C);
  float2* cB   = (float2*)alloc(SZ_C);
  float*  lm   = (float*)cA;    // alias: cA dead after FFT pass 2
  float*  h1   = (float*)cB;    // alias: cB dead after lm extraction
  float*  c2raw = (float*) alloc((size_t)BATCH * 64 * 64 * 64 * 4);
  float*  h2    = (float*) alloc((size_t)BATCH * 64 * 32 * 32 * 4);
  float*  c3raw = (float*) alloc((size_t)BATCH * 128 * 16 * 16 * 4);
  float*  grayP = (float*) alloc((size_t)BATCH * 1024 * 3 * 4);
  float*  lmP   = (float*) alloc((size_t)BATCH * 1024 * 2 * 4);
  float*  eP    = (float*) alloc((size_t)BATCH * 1024 * 7 * 4);
  float*  nP    = (float*) alloc((size_t)BATCH * 1024 * 8 * 4);
  unsigned* hist = (unsigned*) alloc((size_t)BATCH * 50 * 4);
  float*  stats = (float*) alloc((size_t)BATCH * 2 * 4);
  // bf16 packed weights (transposed [N][Kp])
  __bf16* bbWT  = (__bf16*) alloc((size_t)1280 * 32 * 2);
  __bf16* p1WT  = (__bf16*) alloc((size_t)512 * 1280 * 2);
  __bf16* p2WT  = (__bf16*) alloc((size_t)256 * 512 * 2);
  __bf16* fW1T  = (__bf16*) alloc((size_t)256 * 128 * 2);
  __bf16* fW2T  = (__bf16*) alloc((size_t)128 * 256 * 2);
  __bf16* nW1T  = (__bf16*) alloc((size_t)32 * 32 * 2);
  __bf16* nW2T  = (__bf16*) alloc((size_t)64 * 32 * 2);
  __bf16* eW1T  = (__bf16*) alloc((size_t)32 * 32 * 2);
  __bf16* eW2T  = (__bf16*) alloc((size_t)64 * 32 * 2);
  __bf16* fuW1T = (__bf16*) alloc((size_t)256 * 512 * 2);
  __bf16* fuW2T = (__bf16*) alloc((size_t)128 * 256 * 2);
  __bf16* c2Wb  = (__bf16*) alloc((size_t)64 * 288 * 2);
  __bf16* c3Wb  = (__bf16*) alloc((size_t)128 * 576 * 2);
  // bf16 activations
  __bf16* A0     = (__bf16*) alloc((size_t)32 * 32 * 2);
  __bf16* act1   = (__bf16*) alloc((size_t)32 * 1280 * 2);
  __bf16* act2   = (__bf16*) alloc((size_t)32 * 512 * 2);
  __bf16* hfeatB = (__bf16*) alloc((size_t)32 * 128 * 2);
  __bf16* actf   = (__bf16*) alloc((size_t)32 * 256 * 2);
  __bf16* nfeatB = (__bf16*) alloc((size_t)32 * 32 * 2);
  __bf16* actn   = (__bf16*) alloc((size_t)32 * 32 * 2);
  __bf16* efeatB = (__bf16*) alloc((size_t)32 * 32 * 2);
  __bf16* acte   = (__bf16*) alloc((size_t)32 * 32 * 2);
  __bf16* concat = (__bf16*) alloc((size_t)32 * 512 * 2);
  __bf16* actu1  = (__bf16*) alloc((size_t)32 * 256 * 2);
  __bf16* actu2  = (__bf16*) alloc((size_t)32 * 128 * 2);

  // ---- weight packing (independent; front-load) ----
  zero_u32<<<cdiv(BATCH * 50, 256), 256, 0, stream>>>(hist, BATCH * 50);
  pack_wT<<<cdiv(1280*32, 256), 256, 0, stream>>>(bb_w,    bbWT,  3,    1280, 32);
  pack_wT<<<cdiv(512*1280,256), 256, 0, stream>>>(proj_w1, p1WT,  1280, 512,  1280);
  pack_wT<<<cdiv(256*512, 256), 256, 0, stream>>>(proj_w2, p2WT,  512,  256,  512);
  pack_wT<<<cdiv(256*128, 256), 256, 0, stream>>>(f_w1,    fW1T,  128,  256,  128);
  pack_wT<<<cdiv(128*256, 256), 256, 0, stream>>>(f_w2,    fW2T,  256,  128,  256);
  pack_wT<<<cdiv(32*32,   256), 256, 0, stream>>>(n_w1,    nW1T,  6,    32,   32);
  pack_wT<<<cdiv(64*32,   256), 256, 0, stream>>>(n_w2,    nW2T,  32,   64,   32);
  pack_wT<<<cdiv(32*32,   256), 256, 0, stream>>>(e_w1,    eW1T,  5,    32,   32);
  pack_wT<<<cdiv(64*32,   256), 256, 0, stream>>>(e_w2,    eW2T,  32,   64,   32);
  pack_wT<<<cdiv(256*512, 256), 256, 0, stream>>>(fu_w1,   fuW1T, 512,  256,  512);
  pack_wT<<<cdiv(128*256, 256), 256, 0, stream>>>(fu_w2,   fuW2T, 256,  128,  256);
  cvt_f32_bf16<<<cdiv(64*288, 256), 256, 0, stream>>>(c2_w, c2Wb, 64*288);
  cvt_f32_bf16<<<cdiv(128*576,256), 256, 0, stream>>>(c3_w, c3Wb, 128*576);

  // ---- grayscale + channel means ----
  gray_kernel<<<dim3(1024, BATCH), 256, 0, stream>>>(x, gray, grayP);
  finalize_chan<<<BATCH, 256, 0, stream>>>(grayP, A0);

  // ---- FFT branch ----
  fft512<<<dim3(512, BATCH), 256, 0, stream>>>(gray, nullptr, cA, 1);
  fft512<<<dim3(512, BATCH), 256, 0, stream>>>(nullptr, cA, cB, 0);
  lm_kernel<<<dim3(1024, BATCH), 256, 0, stream>>>(cB, lm, lmP);
  finalize_lm<<<BATCH, 256, 0, stream>>>(lmP, stats);
  norm_lm<<<dim3(1024, BATCH), 256, 0, stream>>>(lm, stats);
  conv1_fused<<<dim3(64, 32, BATCH), dim3(16, 16), 0, stream>>>(lm, c1_w, c1_b, bn1_g, bn1_b, h1);
  // conv2: one wave covers all 64 output channels (OCT=4) for its 16 pixels.
  conv_wmma<32, 128, 64, 64, 288, 4><<<dim3(256, 1, BATCH), 32, 0, stream>>>(h1, c2Wb, c2_b, c2raw);
  bn_relu_pool<<<cdiv(BATCH*64*32*32, 256), 256, 0, stream>>>(c2raw, h2, bn2_g, bn2_b, 64, 64);
  // conv3: one wave covers all 128 output channels (OCT=8).
  conv_wmma<64, 32, 128, 16, 576, 8><<<dim3(16, 1, BATCH), 32, 0, stream>>>(h2, c3Wb, c3_b, c3raw);
  conv3_bn_mean<<<dim3(BATCH, 128), 256, 0, stream>>>(c3raw, bn3_g, bn3_b, hfeatB);

  // ---- edge / noise branches ----
  edge_kernel<<<dim3(32, 32, BATCH), dim3(16, 16), 0, stream>>>(gray, eP);
  finalize_edge<<<BATCH, 256, 0, stream>>>(eP, efeatB);
  noise_kernel<<<dim3(32, 32, BATCH), dim3(16, 16), 0, stream>>>(x, gray, nP, hist);
  finalize_noise<<<BATCH, 256, 0, stream>>>(nP, hist, nfeatB);

  // ---- WMMA MLP chain (branch outputs write straight into concat slices) ----
  wmma_gemm<4><<<dim3(20, 2), 32, 0, stream>>>(A0,     32,   bbWT,  32,   bb_b,    act1,   1280, 0,   32,   0);
  wmma_gemm<4><<<dim3(8,  2), 32, 0, stream>>>(act1,   1280, p1WT,  1280, proj_b1, act2,   512,  0,   1280, 1);
  wmma_gemm<4><<<dim3(4,  2), 32, 0, stream>>>(act2,   512,  p2WT,  512,  proj_b2, concat, 512,  0,   512,  1);
  wmma_gemm<4><<<dim3(4,  2), 32, 0, stream>>>(hfeatB, 128,  fW1T,  128,  f_b1,    actf,   256,  0,   128,  1);
  wmma_gemm<4><<<dim3(2,  2), 32, 0, stream>>>(actf,   256,  fW2T,  256,  f_b2,    concat, 512,  256, 256,  1);
  wmma_gemm<2><<<dim3(1,  2), 32, 0, stream>>>(nfeatB, 32,   nW1T,  32,   n_b1,    actn,   32,   0,   32,   1);
  wmma_gemm<4><<<dim3(1,  2), 32, 0, stream>>>(actn,   32,   nW2T,  32,   n_b2,    concat, 512,  384, 32,   1);
  wmma_gemm<2><<<dim3(1,  2), 32, 0, stream>>>(efeatB, 32,   eW1T,  32,   e_b1,    acte,   32,   0,   32,   1);
  wmma_gemm<4><<<dim3(1,  2), 32, 0, stream>>>(acte,   32,   eW2T,  32,   e_b2,    concat, 512,  448, 32,   1);
  wmma_gemm<4><<<dim3(4,  2), 32, 0, stream>>>(concat, 512,  fuW1T, 512,  fu_b1,   actu1,  256,  0,   512,  1);
  wmma_gemm<4><<<dim3(2,  2), 32, 0, stream>>>(actu1,  256,  fuW2T, 256,  fu_b2,   actu2,  128,  0,   256,  1);
  final_logits<<<1, 64, 0, stream>>>(actu2, fu_w3, fu_b3, out);
}
